// BERTModel_87308095193202
// MI455X (gfx1250) — matrix operand order repaired and checked by reading, not compiled, not run on hardware
//
#include <hip/hip_runtime.h>

// ---------------------------------------------------------------------------
// BERT-base forward for MI455X (gfx1250).
// Compute-bound (~1.55 TFLOP vs <0.5 GB unique bytes/layer @ 23.3 TB/s):
// all GEMMs + attention run on v_wmma_f32_16x16x32_bf16 (fp32 accumulate).
// A-tiles are staged into LDS via the Tensor Data Mover (tensor_load_to_lds,
// TENSORcnt), overlapping with the transpose-staging of B-tiles.
// ---------------------------------------------------------------------------

#define BB 16
#define SS 512
#define DD 768
#define HH 12
#define DHH 64
#define LL 12
#define FFD 3072

typedef __attribute__((ext_vector_type(16))) __bf16 v16bf;
typedef __attribute__((ext_vector_type(8)))  float  v8f;
typedef __attribute__((ext_vector_type(4)))  unsigned int u32x4;
typedef __attribute__((ext_vector_type(4)))  int          i32x4;
typedef __attribute__((ext_vector_type(8)))  int          i32x8;

union Frag {
  unsigned short s[16];
  uint4          q[2];
  v16bf          v;
};

__device__ __forceinline__ unsigned short f2bf(float f) {
  return __builtin_bit_cast(unsigned short, (__bf16)f);
}

__device__ __forceinline__ v8f wmma_bf16(const v16bf& a, const v16bf& b, const v8f& c) {
  return __builtin_amdgcn_wmma_f32_16x16x32_bf16(false, a, false, b, (short)0, c, false, false);
}

// TDM issue: toolchains disagree on arity (ROCm7.2 clang-22 = 5 args,
// therock clang-23 = 6 args). The TDM header ships only with the latter.
__device__ __forceinline__ void tdm_load_to_lds(u32x4 g0, i32x8 g1) {
  i32x4 z4 = {};
#if __has_include(<hip/amd_detail/amd_gfx1250_TDM.h>)
  i32x8 z8 = {};
  __builtin_amdgcn_tensor_load_to_lds(g0, g1, z4, z4, z8, 0);
#else
  __builtin_amdgcn_tensor_load_to_lds(g0, g1, z4, z4, 0);
#endif
}

// ---------------------------------------------------------------------------
// Generic bf16 GEMM: C[M,N] = A[M,K] x B[K,N] (+bias,+resid,ReLU)
// Block: 256 threads (8 waves). Tile 128x128, K-step 64 (2 WMMA k-steps).
// Wave (wm,wn) in 4x2 layout owns a 32x64 sub-tile = 2x4 WMMA accumulators.
// A tile: TDM async DMA (wave 0 issues, TENSORcnt).  B tile: manual
// transpose-staging (TDM cannot transpose), overlapped with the DMA.
// ---------------------------------------------------------------------------
template<bool HAS_BIAS, bool RELU, bool RESID, bool OUT_BF16>
__global__ __launch_bounds__(256)
void gemm_bf16_k(const unsigned short* __restrict__ A,
                 const unsigned short* __restrict__ Bw,
                 const float* __restrict__ bias,
                 const float* __restrict__ resid,
                 void* __restrict__ Cout,
                 int M, int N, int K) {
  __shared__ __align__(16) unsigned short As[128][64];
  __shared__ __align__(16) unsigned short Bs[128][64];   // transposed: [n][k]

  const int tid  = threadIdx.x;
  const int lane = tid & 31;
  const int wave = tid >> 5;
  const int wm   = wave >> 1;        // 0..3
  const int wn   = wave & 1;         // 0..1
  const int mblk = blockIdx.y * 128;
  const int nblk = blockIdx.x * 128;

  v8f acc[2][4] = {};

  // ---- constant part of the tensor-DMA descriptor (D# group 1) ----
  // data_size=2B; tensor = K x M (row stride K elems); tile = 64 x 128.
  const unsigned lds_a = (unsigned)(size_t)&As[0][0];
  i32x8 g1;
  g1[0] = 0x00010000;                                   // data_size=1 (2 bytes)
  g1[1] = (int)((K & 0xFFFF) << 16);                    // tensor_dim0 lo16
  g1[2] = (int)(((unsigned)K >> 16) | ((M & 0xFFFF) << 16)); // dim0 hi | dim1 lo
  g1[3] = (int)(((unsigned)M >> 16) | (64u << 16));     // dim1 hi | tile_dim0=64
  g1[4] = 128;                                          // tile_dim1=128, tile_dim2=0
  g1[5] = K;                                            // tensor_dim0_stride lo32
  g1[6] = 0;                                            // stride hi | dim1_stride lo
  g1[7] = 0;

  const int bk  = tid >> 3;            // 0..31
  const int bng = (tid & 7) * 16;      // 0..112

  const int lm    = lane & 15;
  const int khalf = lane >> 4;

  for (int k0 = 0; k0 < K; k0 += 64) {
    // ---- A tile via Tensor Data Mover (async, wave 0 only) ----
    if (wave == 0) {
      const unsigned long long ga =
          (unsigned long long)(size_t)A + ((size_t)mblk * K + k0) * 2ull;
      u32x4 g0;
      g0[0] = 1u;                                    // count=1 (valid D#)
      g0[1] = lds_a;                                 // LDS dest byte offset
      g0[2] = (unsigned)ga;                          // global_addr lo32
      g0[3] = (unsigned)(ga >> 32) | 0x80000000u;    // addr hi | type=2 ("image")
      tdm_load_to_lds(g0, g1);
    }

    // ---- B tile (load row-major along N, store transposed [n][k]) ----
#pragma unroll
    for (int kk2 = 0; kk2 < 2; ++kk2) {
      const uint4* bg =
          (const uint4*)(Bw + (size_t)(k0 + kk2 * 32 + bk) * N + nblk + bng);
      Frag bb;
      bb.q[0] = bg[0];
      bb.q[1] = bg[1];
#pragma unroll
      for (int i = 0; i < 16; ++i) Bs[bng + i][kk2 * 32 + bk] = bb.s[i];
    }
    if (k0 + 64 < K)
      __builtin_prefetch((const void*)(Bw + (size_t)(k0 + 64 + bk) * N + nblk + bng), 0, 1);

    if (wave == 0) __builtin_amdgcn_s_wait_tensorcnt(0);
    __syncthreads();

    // ---- 2 WMMA k-steps over the 64-wide LDS tile ----
#pragma unroll
    for (int ks = 0; ks < 2; ++ks) {
      // A lane layout (16-bit A 16x32): half0 VGPR0-3=K0..7, VGPR4-7=K16..23;
      // half1 = K8..15 / K24..31  -> lo @ 8*khalf, hi @ 16+8*khalf.
      Frag af[2];
#pragma unroll
      for (int mt = 0; mt < 2; ++mt) {
        const int r = wm * 32 + mt * 16 + lm;
        af[mt].q[0] = *(const uint4*)&As[r][ks * 32 + khalf * 8];
        af[mt].q[1] = *(const uint4*)&As[r][ks * 32 + 16 + khalf * 8];
      }
      Frag bf[4];
#pragma unroll
      for (int j = 0; j < 4; ++j) {
        const int c = wn * 64 + j * 16 + lm;
        bf[j].q[0] = *(const uint4*)&Bs[c][ks * 32 + khalf * 8];
        bf[j].q[1] = *(const uint4*)&Bs[c][ks * 32 + 16 + khalf * 8];
      }
#pragma unroll
      for (int mt = 0; mt < 2; ++mt)
#pragma unroll
        for (int j = 0; j < 4; ++j)
          acc[mt][j] = wmma_bf16(af[mt].v, bf[j].v, acc[mt][j]);
    }
    __syncthreads();
  }

  // ---- epilogue: C layout lanes 0-15 -> M=v, lanes 16-31 -> M=8+v, N=lane&15
#pragma unroll
  for (int mt = 0; mt < 2; ++mt) {
#pragma unroll
    for (int j = 0; j < 4; ++j) {
      const int colg = nblk + wn * 64 + j * 16 + lm;
      float cb = 0.0f;
      if (HAS_BIAS) cb = bias[colg];
#pragma unroll
      for (int rv = 0; rv < 8; ++rv) {
        const int rowg = mblk + wm * 32 + mt * 16 + khalf * 8 + rv;
        float val = acc[mt][j][rv] + cb;
        if (RESID) val += resid[(size_t)rowg * N + colg];
        if (RELU)  val = fmaxf(val, 0.0f);
        if (OUT_BF16)
          ((unsigned short*)Cout)[(size_t)rowg * N + colg] = f2bf(val);
        else
          ((float*)Cout)[(size_t)rowg * N + colg] = val;
      }
    }
  }
}

// ---------------------------------------------------------------------------
// Flash-style attention: one block per (64-row query tile, head, batch).
// 4 waves; each wave owns 16 query rows end-to-end (online softmax), so the
// 192x512x512 score tensor never touches HBM. The 64x64 V tile is shared by
// all 4 waves and staged transposed in LDS once per key tile.
// q/k/v/o layout: [b][s][h*64+dh] bf16.
// ---------------------------------------------------------------------------
__global__ __launch_bounds__(128)
void attention_k(const unsigned short* __restrict__ qb,
                 const unsigned short* __restrict__ kb,
                 const unsigned short* __restrict__ vb,
                 unsigned short* __restrict__ ob) {
  const int mtile = blockIdx.x;   // 0..7
  const int h     = blockIdx.y;   // 0..11
  const int b     = blockIdx.z;   // 0..15
  const int tid   = threadIdx.x;
  const int lane  = tid & 31;
  const int wave  = tid >> 5;
  const int lm    = lane & 15;
  const int khalf = lane >> 4;

  __shared__ __align__(16) unsigned short P[4][16][64];
  __shared__ __align__(16) unsigned short Vt[64][64];     // [dh][t]

  const size_t base  = ((size_t)b * SS) * DD + h * DHH;
  const int    mrow0 = mtile * 64 + wave * 16;

  // Q fragments (A operand), DH=64 -> two k-steps of 32
  Frag qf[2];
#pragma unroll
  for (int kk = 0; kk < 2; ++kk) {
    const unsigned short* qp = qb + base + (size_t)(mrow0 + lm) * DD + kk * 32;
    qf[kk].q[0] = *(const uint4*)(qp + khalf * 8);
    qf[kk].q[1] = *(const uint4*)(qp + 16 + khalf * 8);
  }

  v8f  oa[4] = {};
  float rmax[8], rsum[8];
#pragma unroll
  for (int rv = 0; rv < 8; ++rv) { rmax[rv] = -3.0e38f; rsum[rv] = 0.0f; }

  const int vrow = tid >> 1;        // 0..63
  const int vdh0 = (tid & 1) * 32;  // 0 / 32

  for (int tt = 0; tt < SS / 64; ++tt) {
    // ---- cooperatively stage V tile transposed: Vt[dh][t] ----
    {
      const uint4* vp =
          (const uint4*)(vb + base + (size_t)(tt * 64 + vrow) * DD + vdh0);
      Frag va, vb2;
      va.q[0]  = vp[0];  va.q[1]  = vp[1];
      vb2.q[0] = vp[2];  vb2.q[1] = vp[3];
#pragma unroll
      for (int i = 0; i < 16; ++i) Vt[vdh0 + i][vrow]      = va.s[i];
#pragma unroll
      for (int i = 0; i < 16; ++i) Vt[vdh0 + 16 + i][vrow] = vb2.s[i];
    }

    // ---- S = Q K^T for 16x64 strip (K frags straight from global) ----
    v8f sa[4] = {};
#pragma unroll
    for (int kk = 0; kk < 2; ++kk) {
#pragma unroll
      for (int j = 0; j < 4; ++j) {
        const unsigned short* kp =
            kb + base + (size_t)(tt * 64 + j * 16 + lm) * DD + kk * 32 + khalf * 8;
        Frag kf;
        kf.q[0] = *(const uint4*)kp;
        kf.q[1] = *(const uint4*)(kp + 16);
        sa[j] = wmma_bf16(qf[kk].v, kf.v, sa[j]);
      }
    }

    // ---- online softmax (rows in lane-halves; cols across 16 lanes) ----
    float ps[4][8];
#pragma unroll
    for (int rv = 0; rv < 8; ++rv) {
      float tm = -3.0e38f;
#pragma unroll
      for (int j = 0; j < 4; ++j) {
        ps[j][rv] = sa[j][rv] * 0.125f;   // 1/sqrt(64)
        tm = fmaxf(tm, ps[j][rv]);
      }
#pragma unroll
      for (int m = 1; m < 16; m <<= 1) tm = fmaxf(tm, __shfl_xor(tm, m, 32));
      const float nm = fmaxf(rmax[rv], tm);
      const float cr = __expf(rmax[rv] - nm);
      rmax[rv] = nm;
      float lsum = 0.0f;
#pragma unroll
      for (int j = 0; j < 4; ++j) {
        const float p = __expf(ps[j][rv] - nm);
        ps[j][rv] = p;
        lsum += p;
      }
#pragma unroll
      for (int m = 1; m < 16; m <<= 1) lsum += __shfl_xor(lsum, m, 32);
      rsum[rv] = rsum[rv] * cr + lsum;
#pragma unroll
      for (int j = 0; j < 4; ++j) oa[j][rv] *= cr;
    }

    // ---- stage P through LDS to re-layout C->A fragments ----
#pragma unroll
    for (int j = 0; j < 4; ++j)
#pragma unroll
      for (int rv = 0; rv < 8; ++rv)
        P[wave][khalf * 8 + rv][j * 16 + lm] = f2bf(ps[j][rv]);
    __syncthreads();   // orders P (own wave) and Vt (all waves)

    Frag pf[2];
#pragma unroll
    for (int kk = 0; kk < 2; ++kk) {
      pf[kk].q[0] = *(const uint4*)&P[wave][lm][kk * 32 + khalf * 8];
      pf[kk].q[1] = *(const uint4*)&P[wave][lm][kk * 32 + 16 + khalf * 8];
    }

    // ---- O += P V  (V frags as vector LDS reads from Vt) ----
#pragma unroll
    for (int kk = 0; kk < 2; ++kk) {
#pragma unroll
      for (int j = 0; j < 4; ++j) {
        Frag vf;
        vf.q[0] = *(const uint4*)&Vt[j * 16 + lm][kk * 32 + khalf * 8];
        vf.q[1] = *(const uint4*)&Vt[j * 16 + lm][kk * 32 + 16 + khalf * 8];
        oa[j] = wmma_bf16(pf[kk].v, vf.v, oa[j]);
      }
    }
    __syncthreads();   // protect P/Vt before next tile's stores
  }

  // ---- normalize and store (heads concatenated: [b][s][h*64+dh]) ----
#pragma unroll
  for (int j = 0; j < 4; ++j)
#pragma unroll
    for (int rv = 0; rv < 8; ++rv) {
      const float val  = oa[j][rv] / rsum[rv];
      const int   srow = mrow0 + khalf * 8 + rv;
      ob[base + (size_t)srow * DD + j * 16 + lm] = f2bf(val);
    }
}

// ---------------------------------------------------------------------------
// LayerNorm over D=768, one row per block (256 threads, 3 elems/thread).
// ---------------------------------------------------------------------------
template<bool OUT_BF16>
__global__ __launch_bounds__(256)
void layernorm_k(const float* __restrict__ x,
                 const float* __restrict__ g,
                 const float* __restrict__ bta,
                 void* __restrict__ out) {
  __shared__ float red[256];
  const int tid = threadIdx.x;
  const size_t row = blockIdx.x;
  const float* xr = x + row * DD;

  float v0 = xr[tid], v1 = xr[tid + 256], v2 = xr[tid + 512];
  red[tid] = v0 + v1 + v2;
  __syncthreads();
  for (int off = 128; off > 0; off >>= 1) {
    if (tid < off) red[tid] += red[tid + off];
    __syncthreads();
  }
  const float mu = red[0] * (1.0f / DD);
  __syncthreads();

  const float d0 = v0 - mu, d1 = v1 - mu, d2 = v2 - mu;
  red[tid] = d0 * d0 + d1 * d1 + d2 * d2;
  __syncthreads();
  for (int off = 128; off > 0; off >>= 1) {
    if (tid < off) red[tid] += red[tid + off];
    __syncthreads();
  }
  const float rs = rsqrtf(red[0] * (1.0f / DD) + 1e-5f);

  const float y0 = d0 * rs * g[tid]       + bta[tid];
  const float y1 = d1 * rs * g[tid + 256] + bta[tid + 256];
  const float y2 = d2 * rs * g[tid + 512] + bta[tid + 512];
  if (OUT_BF16) {
    unsigned short* o = (unsigned short*)out + row * DD;
    o[tid] = f2bf(y0); o[tid + 256] = f2bf(y1); o[tid + 512] = f2bf(y2);
  } else {
    float* o = (float*)out + row * DD;
    o[tid] = y0; o[tid + 256] = y1; o[tid + 512] = y2;
  }
}

// ---------------------------------------------------------------------------
// Embedding gather + LN_emb fused: one (b,s) row per block.
// ---------------------------------------------------------------------------
__global__ __launch_bounds__(256)
void embed_ln_k(const int* __restrict__ idx,
                const float* __restrict__ tok,
                const float* __restrict__ seg,
                const float* __restrict__ pos,
                const float* __restrict__ g,
                const float* __restrict__ bta,
                float* __restrict__ out) {
  __shared__ float red[256];
  const int tid = threadIdx.x;
  const int row = blockIdx.x;
  const int s   = row % SS;
  const int sg  = (s >= SS / 2 + 1) ? 1 : 0;
  const size_t trow = (size_t)idx[row] * DD;

  float e[3];
#pragma unroll
  for (int i = 0; i < 3; ++i) {
    const int c = tid + i * 256;
    e[i] = tok[trow + c] + seg[(size_t)sg * DD + c] + pos[(size_t)s * DD + c];
  }
  red[tid] = e[0] + e[1] + e[2];
  __syncthreads();
  for (int off = 128; off > 0; off >>= 1) {
    if (tid < off) red[tid] += red[tid + off];
    __syncthreads();
  }
  const float mu = red[0] * (1.0f / DD);
  __syncthreads();
  float d[3];
#pragma unroll
  for (int i = 0; i < 3; ++i) d[i] = e[i] - mu;
  red[tid] = d[0] * d[0] + d[1] * d[1] + d[2] * d[2];
  __syncthreads();
  for (int off = 128; off > 0; off >>= 1) {
    if (tid < off) red[tid] += red[tid + off];
    __syncthreads();
  }
  const float rs = rsqrtf(red[0] * (1.0f / DD) + 1e-5f);
#pragma unroll
  for (int i = 0; i < 3; ++i) {
    const int c = tid + i * 256;
    out[(size_t)row * DD + c] = d[i] * rs * g[c] + bta[c];
  }
}

// ---------------------------------------------------------------------------
// Weight repacks to bf16. QKV: [H][D][DH] -> [D][H*DH].
// ---------------------------------------------------------------------------
__global__ __launch_bounds__(256)
void repack_qkv_k(const float* __restrict__ in, unsigned short* __restrict__ out) {
  const int i = blockIdx.x * 256 + threadIdx.x;
  if (i >= DD * DD) return;
  const int d  = i / DD;
  const int c  = i % DD;
  const int h  = c >> 6;
  const int dh = c & 63;
  out[i] = f2bf(in[((size_t)h * DD + d) * DHH + dh]);
}

__global__ __launch_bounds__(256)
void convert_bf16_k(const float* __restrict__ in, unsigned short* __restrict__ out, int n) {
  for (int i = blockIdx.x * 256 + threadIdx.x; i < n; i += gridDim.x * 256)
    out[i] = f2bf(in[i]);
}

// ---------------------------------------------------------------------------
// Orchestration
// ---------------------------------------------------------------------------
extern "C" void kernel_launch(void* const* d_in, const int* in_sizes, int n_in,
                              void* d_out, int out_size, void* d_ws, size_t ws_size,
                              hipStream_t stream) {
  (void)in_sizes; (void)n_in; (void)out_size; (void)ws_size;

  const int*   idx     = (const int*)d_in[0];
  const float* tok_emb = (const float*)d_in[1];
  const float* seg_emb = (const float*)d_in[2];
  const float* pos_emb = (const float*)d_in[3];
  const float* ln_e_g  = (const float*)d_in[4];
  const float* ln_e_b  = (const float*)d_in[5];
  const float* Wq      = (const float*)d_in[6];
  const float* Wk      = (const float*)d_in[7];
  const float* Wv      = (const float*)d_in[8];
  const float* Wo      = (const float*)d_in[9];
  const float* bo      = (const float*)d_in[10];
  const float* ln1_g   = (const float*)d_in[11];
  const float* ln1_b   = (const float*)d_in[12];
  const float* ln2_g   = (const float*)d_in[13];
  const float* ln2_b   = (const float*)d_in[14];
  const float* W1      = (const float*)d_in[15];
  const float* b1      = (const float*)d_in[16];
  const float* W2      = (const float*)d_in[17];
  const float* b2      = (const float*)d_in[18];
  const float* lnf_g   = (const float*)d_in[19];
  const float* lnf_b   = (const float*)d_in[20];

  const size_t NT = (size_t)BB * SS;          // 8192 rows
  const size_t XB = NT * DD * sizeof(float);  // fp32 activation buffer
  const size_t HB = NT * DD * 2;              // bf16 activation buffer
  const size_t FB = NT * FFD * 2;             // bf16 FFN buffer
  const size_t WB  = (size_t)DD * DD * 2;     // bf16 768x768 weights
  const size_t W1B = (size_t)DD * FFD * 2;    // bf16 768x3072 weights

  char* ws = (char*)d_ws;
  float*          x0   = (float*)(ws);                 // residual stream
  float*          x1   = (float*)(ws + XB);            // temp residual
  unsigned short* hb   = (unsigned short*)(ws + 2 * XB);
  unsigned short* qb   = (unsigned short*)(ws + 2 * XB + HB);
  unsigned short* kb   = (unsigned short*)(ws + 2 * XB + 2 * HB);
  unsigned short* vbuf = (unsigned short*)(ws + 2 * XB + 3 * HB);
  unsigned short* ob   = (unsigned short*)(ws + 2 * XB + 4 * HB);
  unsigned short* fb   = (unsigned short*)(ws + 2 * XB + 5 * HB);
  unsigned short* wq   = (unsigned short*)(ws + 2 * XB + 5 * HB + FB);
  unsigned short* wk = wq + WB / 2;
  unsigned short* wv = wk + WB / 2;
  unsigned short* wo = wv + WB / 2;
  unsigned short* w1 = wo + WB / 2;
  unsigned short* w2 = w1 + W1B / 2;

  const dim3 blk256(256), blk128(128);
  const dim3 gLN((unsigned)NT);
  const dim3 gG768(DD / 128, (unsigned)(NT / 128));    // 6 x 64
  const dim3 gG3072(FFD / 128, (unsigned)(NT / 128));  // 24 x 64
  const dim3 gAtt(SS / 64, HH, BB);                    // 8 x 12 x 16
  const dim3 gRep((DD * DD + 255) / 256);
  const dim3 gCvt(1024);

  // Embedding + LN_emb -> x0 (residual stream)
  embed_ln_k<<<gLN, blk256, 0, stream>>>(idx, tok_emb, seg_emb, pos_emb,
                                         ln_e_g, ln_e_b, x0);

  for (int l = 0; l < LL; ++l) {
    const size_t wqkv_off = (size_t)l * HH * DD * DHH;
    const size_t wo_off   = (size_t)l * DD * DD;
    const size_t wf_off   = (size_t)l * DD * FFD;

    // stage this layer's weights as bf16 (row-major KxN everywhere)
    repack_qkv_k<<<gRep, blk256, 0, stream>>>(Wq + wqkv_off, wq);
    repack_qkv_k<<<gRep, blk256, 0, stream>>>(Wk + wqkv_off, wk);
    repack_qkv_k<<<gRep, blk256, 0, stream>>>(Wv + wqkv_off, wv);
    convert_bf16_k<<<gCvt, blk256, 0, stream>>>(Wo + wo_off, wo, DD * DD);
    convert_bf16_k<<<gCvt, blk256, 0, stream>>>(W1 + wf_off, w1, DD * FFD);
    convert_bf16_k<<<gCvt, blk256, 0, stream>>>(W2 + wf_off, w2, DD * FFD);

    // h = LN1(x0)  (bf16)
    layernorm_k<true><<<gLN, blk256, 0, stream>>>(x0, ln1_g + l * DD, ln1_b + l * DD, hb);

    // q,k,v = h @ W{q,k,v}   (bf16 out)
    gemm_bf16_k<false, false, false, true><<<gG768, blk256, 0, stream>>>(
        hb, wq, nullptr, nullptr, qb, (int)NT, DD, DD);
    gemm_bf16_k<false, false, false, true><<<gG768, blk256, 0, stream>>>(
        hb, wk, nullptr, nullptr, kb, (int)NT, DD, DD);
    gemm_bf16_k<false, false, false, true><<<gG768, blk256, 0, stream>>>(
        hb, wv, nullptr, nullptr, vbuf, (int)NT, DD, DD);

    // o = softmax(qk^T/8) v   (flash, bf16 out, heads concatenated)
    attention_k<<<gAtt, blk128, 0, stream>>>(qb, kb, vbuf, ob);

    // x1 = x0 + o @ Wo + bo   (fp32)
    gemm_bf16_k<true, false, true, false><<<gG768, blk256, 0, stream>>>(
        ob, wo, bo + l * DD, x0, x1, (int)NT, DD, DD);

    // h = LN2(x1)
    layernorm_k<true><<<gLN, blk256, 0, stream>>>(x1, ln2_g + l * DD, ln2_b + l * DD, hb);

    // ff = relu(h @ W1 + b1)  (bf16)
    gemm_bf16_k<true, true, false, true><<<gG3072, blk256, 0, stream>>>(
        hb, w1, b1 + l * FFD, nullptr, fb, (int)NT, FFD, DD);

    // x0 = x1 + ff @ W2 + b2  (fp32) -> residual stream stays in x0
    gemm_bf16_k<true, false, true, false><<<gG768, blk256, 0, stream>>>(
        fb, w2, b2 + l * DD, x1, x0, (int)NT, DD, FFD);
  }

  // final LN -> d_out (fp32)
  layernorm_k<false><<<gLN, blk256, 0, stream>>>(x0, lnf_g, lnf_b, (float*)d_out);
}
